// Encoder_36644660970131
// MI455X (gfx1250) — matrix operand order repaired and checked by reading, not compiled, not run on hardware
//
#include <hip/hip_runtime.h>

// ---------------------------------------------------------------------------
// MI455X (gfx1250) encoder: f16 WMMA GEMMs with f32 accumulation.
//   L=6 D=1024 H=16 DK=64 S=512 B=8 DFF=4096
// ~0.91 TFLOP of GEMM work; 822MB attention-prob output -> memory floor
// ~50-60us @ 23.3TB/s.  Matmuls use v_wmma_f32_16x16x32_f16 (8x K/instr vs
// the only f32 WMMA shape 16x16x4).  128x64 block tile, 32x32 wave tile:
// 4 wmma per 4 LDS fragment loads per K-step.
// ---------------------------------------------------------------------------

typedef _Float16 half_t;
typedef __attribute__((ext_vector_type(16))) _Float16 v16h;
typedef __attribute__((ext_vector_type(8)))  _Float16 v8h;
typedef __attribute__((ext_vector_type(8)))  float    v8f;
typedef __attribute__((ext_vector_type(4)))  float    v4f;

#define DEV_INLINE __device__ __forceinline__

// ---- wave32 / block reductions (blockDim.x == 256 -> 8 waves) -------------
DEV_INLINE float wave_sum(float v) {
#pragma unroll
    for (int o = 16; o > 0; o >>= 1) v += __shfl_xor(v, o, 32);
    return v;
}
DEV_INLINE float wave_max(float v) {
#pragma unroll
    for (int o = 16; o > 0; o >>= 1) v = fmaxf(v, __shfl_xor(v, o, 32));
    return v;
}
DEV_INLINE float block_sum(float v, float* red) {
    const int lane = threadIdx.x & 31, w = threadIdx.x >> 5;
    v = wave_sum(v);
    __syncthreads();                 // protect red across consecutive calls
    if (lane == 0) red[w] = v;
    __syncthreads();
    float t = red[0];
#pragma unroll
    for (int i = 1; i < 8; ++i) t += red[i];
    return t;
}
DEV_INLINE float block_max(float v, float* red) {
    const int lane = threadIdx.x & 31, w = threadIdx.x >> 5;
    v = wave_max(v);
    __syncthreads();
    if (lane == 0) red[w] = v;
    __syncthreads();
    float t = red[0];
#pragma unroll
    for (int i = 1; i < 8; ++i) t = fmaxf(t, red[i]);
    return t;
}

// ---------------------------------------------------------------------------
// Batched WMMA GEMM:  C[z] = alpha * A[z] x B[z] (+bias) (+C_old if ACC)
//   A: [M,K] f16 row-major (lda). B: NN -> [K,N] (ldb); BT -> [N,K] (ldb).
//   Batch z decomposed as (zb, zh) = (z / batchH, z % batchH) with two-level
//   strides so [B,H,S,*] attention slices address correctly.
// Requirements (satisfied by this model): M%128==0 (grid.y), N%64==0
// (grid.x), K%32==0.  Block = 256 threads = 8 wave32s; block tile 128x64,
// wave tile 32x32 -> 4 x v_wmma_f32_16x16x32_f16 per wave per K-step.
// ---------------------------------------------------------------------------
static constexpr int BM = 128, BN = 64, BKs = 32, LK = BKs + 8;

template <typename OutT, bool BT, bool RELU, bool ACC>
__global__ void __launch_bounds__(256)
gemm_wmma(const half_t* __restrict__ A, int lda, long long sA0, long long sA1,
          const half_t* __restrict__ B, int ldb, long long sB0, long long sB1,
          OutT* __restrict__ C, int ldc, long long sC0, long long sC1,
          int K, int batchH, float alpha, const float* __restrict__ bias)
{
    __shared__ half_t As[BM][LK];
    __shared__ half_t Bs[BN][LK];   // stored transposed: [n][k]

    const int z  = blockIdx.z;
    const int zb = z / batchH;
    const int zh = z - zb * batchH;
    A += zb * sA0 + zh * sA1;
    B += zb * sB0 + zh * sB1;
    C += zb * sC0 + zh * sC1;

    const int bm   = blockIdx.y * BM;
    const int bn   = blockIdx.x * BN;
    const int tid  = threadIdx.x;
    const int lane = tid & 31;
    const int w    = tid >> 5;
    const int wm   = (w & 3) * 32;   // wave row offset in tile (0,32,64,96)
    const int wn   = (w >> 2) * 32;  // wave col offset (0,32)

    // staging indices
    const int ar  = tid >> 1;        // 0..127 A row
    const int ac  = (tid & 1) * 16;  // 0,16
    const int br  = tid >> 2;        // 0..63  Bt row (BT path)
    const int bc  = (tid & 3) * 8;   // 0,8,16,24
    const int bk  = tid >> 3;        // 0..31 k-row for NN B staging
    const int bn8 = (tid & 7) * 8;   // 0..56 (coalesced 128B per 8 lanes)

    const int fr = lane & 15;
    const int hi = lane >> 4;        // 0 or 1

    v8f acc00 = {}, acc01 = {}, acc10 = {}, acc11 = {};

    for (int k0 = 0; k0 < K; k0 += BKs) {
        // A tile 128x32: each thread moves 16 halfs (two b128 transfers)
        {
            const long long g = (long long)(bm + ar) * lda + (k0 + ac);
            *(v8h*)&As[ar][ac]     = *(const v8h*)&A[g];
            *(v8h*)&As[ar][ac + 8] = *(const v8h*)&A[g + 8];
        }
        if (BT) {
            *(v8h*)&Bs[br][bc] = *(const v8h*)&B[(long long)(bn + br) * ldb + (k0 + bc)];
        } else {
            const v8h t = *(const v8h*)&B[(long long)(k0 + bk) * ldb + (bn + bn8)];
#pragma unroll
            for (int i = 0; i < 8; ++i) Bs[bn8 + i][bk] = t[i];
        }
        if (k0 + BKs < K) {          // global_prefetch_b8 next K tile
            __builtin_prefetch(&A[(long long)(bm + ar) * lda + (k0 + BKs + ac)], 0, 1);
            if (BT) __builtin_prefetch(&B[(long long)(bn + br) * ldb + (k0 + BKs + bc)], 0, 1);
            else    __builtin_prefetch(&B[(long long)(k0 + BKs + bk) * ldb + (bn + bn8)], 0, 1);
        }
        __syncthreads();

        // A frag (ISA 16-bit 16x32): lanes 0-15 -> K 0..7 & 16..23,
        //                            lanes16-31 -> K 8..15 & 24..31.
        v16h a0, a1, b0, b1;
        const int ka = hi * 8;
#pragma unroll
        for (int i = 0; i < 8; ++i) {
            a0[i]     = As[wm + fr][ka + i];
            a0[i + 8] = As[wm + fr][ka + 16 + i];
            a1[i]     = As[wm + 16 + fr][ka + i];
            a1[i + 8] = As[wm + 16 + fr][ka + 16 + i];
        }
        // B frag (32x16): same K order as A: lanes 0-15 -> K 0..7 & 16..23,
        //                 lanes 16-31 -> K 8..15 & 24..31.
        const int kb = hi * 8;
#pragma unroll
        for (int i = 0; i < 8; ++i) {
            b0[i]     = Bs[wn + fr][kb + i];
            b0[i + 8] = Bs[wn + fr][kb + 16 + i];
            b1[i]     = Bs[wn + 16 + fr][kb + i];
            b1[i + 8] = Bs[wn + 16 + fr][kb + 16 + i];
        }
        acc00 = __builtin_amdgcn_wmma_f32_16x16x32_f16(false, a0, false, b0,
                                                       (short)0, acc00, false, false);
        acc01 = __builtin_amdgcn_wmma_f32_16x16x32_f16(false, a0, false, b1,
                                                       (short)0, acc01, false, false);
        acc10 = __builtin_amdgcn_wmma_f32_16x16x32_f16(false, a1, false, b0,
                                                       (short)0, acc10, false, false);
        acc11 = __builtin_amdgcn_wmma_f32_16x16x32_f16(false, a1, false, b1,
                                                       (short)0, acc11, false, false);
        __syncthreads();
    }

    // C frag: VGPR r -> M = r + 8*hi, N = lane&15
    const int row0 = bm + wm + hi * 8;
    const int col0 = bn + wn + fr;
    auto epi = [&](const v8f& acc, int row, int col) {
#pragma unroll
        for (int r = 0; r < 8; ++r) {
            float v = alpha * acc[r];
            if (bias) v += bias[col];
            if (RELU) v = fmaxf(v, 0.0f);
            const long long idx = (long long)(row + r) * ldc + col;
            if (ACC) v += (float)C[idx];
            C[idx] = (OutT)v;
        }
    };
    epi(acc00, row0,      col0);
    epi(acc01, row0,      col0 + 16);
    epi(acc10, row0 + 16, col0);
    epi(acc11, row0 + 16, col0 + 16);
}

// ---------------------------------------------------------------------------
// LayerNorm over D=1024, one row per block, 4 elems/thread (float4).
// Optionally writes an f16 copy for the following WMMA GEMM.
// ---------------------------------------------------------------------------
__global__ void __launch_bounds__(256)
layernorm_kernel(const float* __restrict__ x, const float* __restrict__ g,
                 const float* __restrict__ bb, float* __restrict__ y32,
                 half_t* __restrict__ y16)
{
    const int row = blockIdx.x, tid = threadIdx.x;
    const long long base = (long long)row * 1024 + tid * 4;
    const v4f xv = *(const v4f*)(x + base);
    float s  = xv[0] + xv[1] + xv[2] + xv[3];
    float sq = xv[0]*xv[0] + xv[1]*xv[1] + xv[2]*xv[2] + xv[3]*xv[3];
    __shared__ float red[8];
    const float mean = block_sum(s,  red) * (1.0f / 1024.0f);
    const float var  = block_sum(sq, red) * (1.0f / 1024.0f) - mean * mean;
    const float rstd = rsqrtf(var + 1e-5f);
    const v4f gv = *(const v4f*)(g  + tid * 4);
    const v4f bv = *(const v4f*)(bb + tid * 4);
    v4f o;
#pragma unroll
    for (int i = 0; i < 4; ++i) o[i] = (xv[i] - mean) * rstd * gv[i] + bv[i];
    if (y32) *(v4f*)(y32 + base) = o;
    if (y16) {
        half_t* yr = y16 + base;
#pragma unroll
        for (int i = 0; i < 4; ++i) yr[i] = (half_t)o[i];
    }
}

// z = att2 + qin;  h += LN(z)*g + b    (AttSubLayerv2 post-norm + residual)
__global__ void __launch_bounds__(256)
add_ln_res_kernel(const float* __restrict__ att2, const float* __restrict__ qin,
                  float* __restrict__ h, const float* __restrict__ g,
                  const float* __restrict__ bb)
{
    const int row = blockIdx.x, tid = threadIdx.x;
    const long long base = (long long)row * 1024 + tid * 4;
    const v4f a = *(const v4f*)(att2 + base);
    const v4f q = *(const v4f*)(qin  + base);
    v4f zv;
#pragma unroll
    for (int i = 0; i < 4; ++i) zv[i] = a[i] + q[i];
    float s  = zv[0] + zv[1] + zv[2] + zv[3];
    float sq = zv[0]*zv[0] + zv[1]*zv[1] + zv[2]*zv[2] + zv[3]*zv[3];
    __shared__ float red[8];
    const float mean = block_sum(s,  red) * (1.0f / 1024.0f);
    const float var  = block_sum(sq, red) * (1.0f / 1024.0f) - mean * mean;
    const float rstd = rsqrtf(var + 1e-5f);
    const v4f gv = *(const v4f*)(g  + tid * 4);
    const v4f bv = *(const v4f*)(bb + tid * 4);
    v4f hv = *(const v4f*)(h + base);
#pragma unroll
    for (int i = 0; i < 4; ++i) hv[i] += (zv[i] - mean) * rstd * gv[i] + bv[i];
    *(v4f*)(h + base) = hv;
}

// Masked softmax over S=512, one (b,h,q) row per block, in place in d_out,
// plus an f16 copy of p for the p@V WMMA GEMM.
__global__ void __launch_bounds__(256)
softmax_mask_kernel(float* __restrict__ scores, const int* __restrict__ mask,
                    half_t* __restrict__ p16, int S, int HSdim)
{
    const int row = blockIdx.x;             // 0 .. B*H*S-1
    const int b   = row / HSdim;
    const int q   = row % S;
    const int tid = threadIdx.x;
    float*       srow = scores + (long long)row * S;
    const int*   mrow = mask + ((long long)b * S + q) * S;
    half_t*      prow = p16 + (long long)row * S;

    float v[2];
#pragma unroll
    for (int i = 0; i < 2; ++i) {
        const int k = tid + i * 256;
        v[i] = mrow[k] ? -1.0e9f : srow[k];
    }
    __shared__ float red[8];
    const float mx = block_max(fmaxf(v[0], v[1]), red);
    float e[2], s = 0.0f;
#pragma unroll
    for (int i = 0; i < 2; ++i) { e[i] = __expf(v[i] - mx); s += e[i]; }
    const float inv = 1.0f / block_sum(s, red);
#pragma unroll
    for (int i = 0; i < 2; ++i) {
        const int   k = tid + i * 256;
        const float o = e[i] * inv;
        srow[k] = o;
        prow[k] = (half_t)o;
    }
}

__global__ void __launch_bounds__(256)
f32_to_f16_kernel(const float* __restrict__ x, half_t* __restrict__ y, long long n)
{
    long long i = (long long)blockIdx.x * blockDim.x + threadIdx.x;
    const long long stride = (long long)gridDim.x * blockDim.x;
    for (; i < n; i += stride) y[i] = (half_t)x[i];
}

// ---------------------------------------------------------------------------
extern "C" void kernel_launch(void* const* d_in, const int* in_sizes, int n_in,
                              void* d_out, int out_size, void* d_ws, size_t ws_size,
                              hipStream_t stream)
{
    (void)in_sizes; (void)n_in; (void)out_size; (void)ws_size;
    constexpr int L = 6, D = 1024, H = 16, DK = 64, S = 512, B = 8, DFF = 4096;
    constexpr int HD = H * DK;                       // 1024
    constexpr int M  = B * S;                        // 4096 rows
    constexpr long long ATT = (long long)B * H * S * S;  // attn elems / layer

    const float* x        = (const float*)d_in[0];
    const int*   mask     = (const int*)  d_in[1];
    const float* relation = (const float*)d_in[2];
    const float* Wq   = (const float*)d_in[3];
    const float* Wk   = (const float*)d_in[4];
    const float* Wv   = (const float*)d_in[5];
    const float* Wo   = (const float*)d_in[6];
    const float* bo   = (const float*)d_in[7];
    const float* lng  = (const float*)d_in[8];
    const float* lnb  = (const float*)d_in[9];
    const float* ln2g = (const float*)d_in[10];
    const float* ln2b = (const float*)d_in[11];
    const float* W1   = (const float*)d_in[12];
    const float* b1   = (const float*)d_in[13];
    const float* W2   = (const float*)d_in[14];
    const float* b2   = (const float*)d_in[15];
    const float* lnfg = (const float*)d_in[16];
    const float* lnfb = (const float*)d_in[17];

    float* out   = (float*)d_out;
    float* attn0 = out + (long long)M * D;           // attn slab in d_out

    // ---- workspace arenas (~222 MB) ----
    char* p = (char*)d_ws;
    auto arena = [&](long long bytes) -> char* {
        char* r = p; p += (bytes + 255) & ~255LL; return r;
    };
    float*  h     = (float*) arena((long long)M * D * 4);
    float*  qin32 = (float*) arena((long long)M * D * 4);
    float*  att2  = (float*) arena((long long)M * D * 4);
    half_t* qin16 = (half_t*)arena((long long)M * D * 2);
    half_t* Q16   = (half_t*)arena((long long)M * HD * 2);
    half_t* K16   = (half_t*)arena((long long)M * HD * 2);
    half_t* V16   = (half_t*)arena((long long)M * HD * 2);
    half_t* att16 = (half_t*)arena((long long)M * HD * 2);
    half_t* t116  = (half_t*)arena((long long)M * DFF * 2);
    half_t* qk16  = (half_t*)arena(ATT * 2);         // reused as p16 post-softmax
    half_t* rel16 = (half_t*)arena((long long)B * S * S * 2);
    half_t* wq16  = (half_t*)arena((long long)D * HD * 2);
    half_t* wk16  = (half_t*)arena((long long)D * HD * 2);
    half_t* wv16  = (half_t*)arena((long long)D * HD * 2);
    half_t* wo16  = (half_t*)arena((long long)HD * D * 2);
    half_t* w116  = (half_t*)arena((long long)D * DFF * 2);
    half_t* w216  = (half_t*)arena((long long)DFF * D * 2);

    const dim3 blk(256);

    hipMemcpyAsync(h, x, (size_t)M * D * 4, hipMemcpyDeviceToDevice, stream);
    f32_to_f16_kernel<<<1024, blk, 0, stream>>>(relation, rel16, (long long)B * S * S);

    for (int l = 0; l < L; ++l) {
        const float* Wq_l  = Wq + (long long)l * D * HD;
        const float* Wk_l  = Wk + (long long)l * D * HD;
        const float* Wv_l  = Wv + (long long)l * D * HD;
        const float* Wo_l  = Wo + (long long)l * HD * D;
        const float* bo_l  = bo + l * D;
        const float* lng_l = lng + l * D;
        const float* lnb_l = lnb + l * D;
        const float* g2_l  = ln2g + l * D;
        const float* b2n_l = ln2b + l * D;
        const float* W1_l  = W1 + (long long)l * D * DFF;
        const float* b1_l  = b1 + l * DFF;
        const float* W2_l  = W2 + (long long)l * DFF * D;
        const float* b2_l  = b2 + l * D;
        float* attn_l = attn0 + (long long)l * ATT;

        // q_in = LN(h)   (f32 for residual, f16 for GEMM)
        layernorm_kernel<<<M, blk, 0, stream>>>(h, lng_l, lnb_l, qin32, qin16);

        // per-layer weight conversion (stays resident in 192MB L2)
        f32_to_f16_kernel<<<1024, blk, 0, stream>>>(Wq_l, wq16, (long long)D * HD);
        f32_to_f16_kernel<<<1024, blk, 0, stream>>>(Wk_l, wk16, (long long)D * HD);
        f32_to_f16_kernel<<<1024, blk, 0, stream>>>(Wv_l, wv16, (long long)D * HD);
        f32_to_f16_kernel<<<1024, blk, 0, stream>>>(Wo_l, wo16, (long long)HD * D);
        f32_to_f16_kernel<<<1024, blk, 0, stream>>>(W1_l, w116, (long long)D * DFF);
        f32_to_f16_kernel<<<1024, blk, 0, stream>>>(W2_l, w216, (long long)DFF * D);

        // Q/K/V = q_in x W   [4096 x 1024 x 1024], f16 out
        gemm_wmma<half_t, false, false, false><<<dim3(HD / BN, M / BM, 1), blk, 0, stream>>>(
            qin16, D, 0, 0, wq16, HD, 0, 0, Q16, HD, 0, 0, D, 1, 1.0f, nullptr);
        gemm_wmma<half_t, false, false, false><<<dim3(HD / BN, M / BM, 1), blk, 0, stream>>>(
            qin16, D, 0, 0, wk16, HD, 0, 0, K16, HD, 0, 0, D, 1, 1.0f, nullptr);
        gemm_wmma<half_t, false, false, false><<<dim3(HD / BN, M / BM, 1), blk, 0, stream>>>(
            qin16, D, 0, 0, wv16, HD, 0, 0, V16, HD, 0, 0, D, 1, 1.0f, nullptr);

        // qk = Q x K^T per (b,h): 128 batches of [512 x 512 x 64], B is N-major
        gemm_wmma<half_t, true, false, false><<<dim3(S / BN, S / BM, B * H), blk, 0, stream>>>(
            Q16,  HD, (long long)S * HD, DK,
            K16,  HD, (long long)S * HD, DK,
            qk16, S,  (long long)H * S * S, (long long)S * S,
            DK, H, 1.0f, nullptr);

        // scores = (qk x relation) * 1/sqrt(DK), written into d_out attn slice
        gemm_wmma<float, false, false, false><<<dim3(S / BN, (H * S) / BM, B), blk, 0, stream>>>(
            qk16,   S, (long long)H * S * S, 0,
            rel16,  S, (long long)S * S,     0,
            attn_l, S, (long long)H * S * S, 0,
            S, 1, 0.125f, nullptr);

        // p = softmax(mask ? NEG : scores), in place; f16 copy reuses qk16
        softmax_mask_kernel<<<B * H * S, blk, 0, stream>>>(attn_l, mask, qk16, S, H * S);

        // att = p x V per (b,h): [512 x 64 x 512]; writes directly in [B,S,H*DK]
        gemm_wmma<half_t, false, false, false><<<dim3(1, S / BM, B * H), blk, 0, stream>>>(
            qk16,  S,  (long long)H * S * S, (long long)S * S,
            V16,   HD, (long long)S * HD,    DK,
            att16, HD, (long long)S * HD,    DK,
            S, H, 1.0f, nullptr);

        // att2 = att x Wo + bo
        gemm_wmma<float, false, false, false><<<dim3(D / BN, M / BM, 1), blk, 0, stream>>>(
            att16, HD, 0, 0, wo16, D, 0, 0, att2, D, 0, 0, HD, 1, 1.0f, bo_l);

        // h += LN(att2 + q_in; g2, b2)
        add_ln_res_kernel<<<M, blk, 0, stream>>>(att2, qin32, h, g2_l, b2n_l);

        // f = LN(h) (f16 only, reuse qin16)
        layernorm_kernel<<<M, blk, 0, stream>>>(h, lng_l, lnb_l, nullptr, qin16);

        // t1 = relu(f x W1 + b1)   [4096 x 4096 x 1024], f16 out
        gemm_wmma<half_t, false, true, false><<<dim3(DFF / BN, M / BM, 1), blk, 0, stream>>>(
            qin16, D, 0, 0, w116, DFF, 0, 0, t116, DFF, 0, 0, D, 1, 1.0f, b1_l);

        // h += t1 x W2 + b2        [4096 x 1024 x 4096], accumulate into h
        gemm_wmma<float, false, false, true><<<dim3(D / BN, M / BM, 1), blk, 0, stream>>>(
            t116, DFF, 0, 0, w216, D, 0, 0, h, D, 0, 0, DFF, 1, 1.0f, b2_l);
    }

    // out = LN(h; lnf)
    layernorm_kernel<<<M, blk, 0, stream>>>(h, lnfg, lnfb, out, nullptr);
}
